// RWKV7Attention_62912680952658
// MI455X (gfx1250) — matrix operand order repaired and checked by reading, not compiled, not run on hardware
//
#include <hip/hip_runtime.h>
#include <stdint.h>

// ---------------- problem constants ----------------
#define B_   4
#define T_   2048
#define H_   1024
#define HD_  64
#define NH_  16
#define M_   (B_*T_)                 // 8192 token rows
#define EPS_ (HD_*1e-5f)
#define DECAY_ (-0.6065306597126334f)

typedef unsigned short u16;
typedef __attribute__((ext_vector_type(16))) __bf16 v16bf;
typedef __attribute__((ext_vector_type(8)))  float  v8f;
typedef __attribute__((ext_vector_type(4)))  unsigned int v4u;
typedef __attribute__((ext_vector_type(8)))  int  v8i;
typedef __attribute__((ext_vector_type(4)))  int  v4i;

__device__ __forceinline__ u16 f2bf(float f) {
  union { float f; uint32_t u; } c; c.f = f;
  uint32_t u = c.u;
  u += 0x7FFFu + ((u >> 16) & 1u);      // round-to-nearest-even
  return (u16)(u >> 16);
}
__device__ __forceinline__ float sigm(float x) { return 1.0f / (1.0f + __expf(-x)); }

// ---------------- token shift + interpolation -> 6 bf16 activation mats ----
__global__ __launch_bounds__(256) void k_token_shift(
    const float* __restrict__ x,
    const float* __restrict__ mr, const float* __restrict__ mw,
    const float* __restrict__ mk, const float* __restrict__ mv,
    const float* __restrict__ ma, const float* __restrict__ mg,
    u16* __restrict__ xr, u16* __restrict__ xw, u16* __restrict__ xk,
    u16* __restrict__ xv, u16* __restrict__ xa, u16* __restrict__ xg) {
  long idx = (long)blockIdx.x * blockDim.x + threadIdx.x;
  if (idx >= (long)M_ * H_) return;
  int  h  = (int)(idx % H_);
  long bt = idx / H_;
  int  t  = (int)(bt % T_);
  float xc = x[idx];
  float xp = (t == 0) ? 0.f : x[idx - H_];
  float d  = xp - xc;
  xr[idx] = f2bf(xc + d * mr[h]);
  xw[idx] = f2bf(xc + d * mw[h]);
  xk[idx] = f2bf(xc + d * mk[h]);
  xv[idx] = f2bf(xc + d * mv[h]);
  xa[idx] = f2bf(xc + d * ma[h]);
  xg[idx] = f2bf(xc + d * mg[h]);
}

// ---------------- f32 -> bf16 convert (op: 0 none, 1 tanh, 2 sigmoid) ------
__global__ __launch_bounds__(256) void k_cvt(
    const float* __restrict__ src, u16* __restrict__ dst, long n, int op) {
  long i = (long)blockIdx.x * blockDim.x + threadIdx.x;
  if (i >= n) return;
  float v = src[i];
  if      (op == 1) v = tanhf(v);
  else if (op == 2) v = sigm(v);
  dst[i] = f2bf(v);
}

// ---------------- bf16 WMMA GEMM: C[M,N] = A[M,K] @ W[K,N] -----------------
// Block tile 128x128, K-step 32. 8 waves (4x2); wave tile 32x64 = 2x4 WMMA
// fragments of 16x16, each K-step issues 8 x v_wmma_f32_16x16x32_bf16.
// A tile is DMA'd by the Tensor Data Mover (tensor_load_to_lds) with hardware
// LDS padding (64B rows + 16B pad == 40-ushort stride); W tile is staged
// transposed by the waves with b128 global loads.
#define BM  128
#define BN  128
#define BK  32
#define BKP 40   // padded LDS row stride (bf16 units): 64B data + 16B pad

__global__ __launch_bounds__(256, 1) void k_gemm_bf16(
    const u16* __restrict__ A, const u16* __restrict__ W,
    float* __restrict__ C, int N, int K) {
  __shared__ u16 As[BM][BKP];       // A tile, row-major [m][k], TDM-filled
  __shared__ u16 Ws[BN][BKP];       // W tile, transposed [n][k]

  const int tid   = threadIdx.x;
  const int lane  = tid & 31;
  const int wave  = tid >> 5;
  const int g     = lane >> 4;      // lane group (0/1) within wave32
  const int l16   = lane & 15;
  const int waveM = wave & 3;       // 4 waves along M
  const int waveN = wave >> 2;      // 2 waves along N
  const int mBase = blockIdx.y * BM;
  const int nBase = blockIdx.x * BN;

  v8f acc[2][4];
  #pragma unroll
  for (int i = 0; i < 2; i++)
    #pragma unroll
    for (int j = 0; j < 4; j++)
      #pragma unroll
      for (int e = 0; e < 8; e++) acc[i][j][e] = 0.f;

  const int kTiles = K / BK;
  for (int kt = 0; kt < kTiles; ++kt) {
    const int k0 = kt * BK;

    // ---- A tile via Tensor Data Mover (wave 0, uniform SGPR descriptor) ---
    if (wave == 0) {
      uint64_t ga  = (uint64_t)(uintptr_t)(A + (long)mBase * K + k0);
      uint32_t lds = (uint32_t)(uintptr_t)(&As[0][0]);
      v4u g0;
      g0[0] = 1u;                                    // count=1, no gather
      g0[1] = lds;                                   // lds_addr (bytes)
      g0[2] = (uint32_t)ga;                          // global_addr[31:0]
      g0[3] = (uint32_t)((ga >> 32) & 0x01FFFFFFu)   // global_addr[56:32]
              | (2u << 30);                          // type = 2 ("image")
      v8i g1;
      g1[0] = (int)((1u << 16)    // data_size = 1 -> 2-byte elements
                  | (1u << 20)    // pad_enable
                  | (3u << 22)    // pad_interval: 16 DWORDs (64B) per row
                  | (3u << 25));  // pad_amount: 4 DWORDs (16B) pad
      g1[1] = (int)((unsigned)BK << 16);   // tensor_dim0 = 32  (tile-sized)
      g1[2] = (int)((unsigned)BM << 16);   // tensor_dim1 = 128 (tile-sized)
      g1[3] = (int)((unsigned)BK << 16);   // tile_dim0 = 32
      g1[4] = BM;                          // tile_dim1 = 128, tile_dim2 = 0
      g1[5] = K;                           // tensor_dim0_stride = K elems
      g1[6] = 0; g1[7] = 0;
      v4i z4 = {0, 0, 0, 0};
      v8i z8 = {0, 0, 0, 0, 0, 0, 0, 0};
      __builtin_amdgcn_tensor_load_to_lds(g0, g1, z4, z4, z8, 0);
    }

    // ---- W tile transposed: Ws[n][k] = W[k0+k][nBase+n], b128 loads ------
    #pragma unroll
    for (int i = 0; i < 2; i++) {
      int oct = tid + i * 256;           // 0..511 octets of 8 bf16
      int k   = oct >> 4;                // 0..31
      int no  = (oct & 15) * 8;          // 0..120
      int gn  = nBase + no;
      union { uint4 q; u16 h[8]; } d;
      if (gn + 8 <= N) {
        d.q = *(const uint4*)(W + (long)(k0 + k) * N + gn);
      } else {
        #pragma unroll
        for (int e = 0; e < 8; e++)
          d.h[e] = (gn + e < N) ? W[(long)(k0 + k) * N + gn + e] : (u16)0;
      }
      #pragma unroll
      for (int e = 0; e < 8; e++) Ws[no + e][k] = d.h[e];
    }
    if (kt + 1 < kTiles)  // hint next W K-tile into cache (global_prefetch)
      __builtin_prefetch((const void*)(W + (long)(k0 + BK + (tid >> 3)) * N + nBase), 0, 1);

    if (wave == 0) __builtin_amdgcn_s_wait_tensorcnt(0);
    __syncthreads();

    // ---- build fragments per ISA layout ----------------------------------
    union FR { v16bf v; uint32_t u[8]; };
    FR afr[2], bfr[4];
    #pragma unroll
    for (int fm = 0; fm < 2; fm++) {
      int row = waveM * 32 + fm * 16 + l16;   // M = lane%16
      const uint32_t* p = (const uint32_t*)&As[row][0];
      #pragma unroll
      for (int v = 0; v < 8; v++) {
        int kk = ((v < 4) ? 0 : 16) + g * 8 + (v & 3) * 2;  // A 16-bit layout
        afr[fm].u[v] = p[kk >> 1];
      }
    }
    #pragma unroll
    for (int fn = 0; fn < 4; fn++) {
      int col = waveN * 64 + fn * 16 + l16;   // N = lane%16
      const uint32_t* p = (const uint32_t*)&Ws[col][0];
      #pragma unroll
      for (int v = 0; v < 8; v++) {
        int kk = ((v < 4) ? 0 : 16) + g * 8 + (v & 3) * 2;  // B 16-bit layout
        bfr[fn].u[v] = p[kk >> 1];
      }
    }
    #pragma unroll
    for (int fm = 0; fm < 2; fm++)
      #pragma unroll
      for (int fn = 0; fn < 4; fn++)
        acc[fm][fn] = __builtin_amdgcn_wmma_f32_16x16x32_bf16(
            false, afr[fm].v, false, bfr[fn].v, (short)0, acc[fm][fn], false, false);
    __syncthreads();
  }

  // store: C/D layout — VGPR r holds M = r + 8*(lane/16), N = lane%16
  #pragma unroll
  for (int fm = 0; fm < 2; fm++)
    #pragma unroll
    for (int fn = 0; fn < 4; fn++) {
      int col = nBase + waveN * 64 + fn * 16 + l16;
      if (col < N) {
        #pragma unroll
        for (int r = 0; r < 8; r++) {
          int row = mBase + waveM * 32 + fm * 16 + r + 8 * g;
          C[(long)row * N + col] = acc[fm][fn][r];
        }
      }
    }
}

// ---------------- post-LoRA elementwise: w, a, v --------------------------
__global__ __launch_bounds__(256) void k_post(
    float* __restrict__ wpre,          // in: xw lora out -> out: log-decay w
    float* __restrict__ apre,          // in: xa lora out -> out: a (sigmoid)
    const float* __restrict__ vt,      // xv lora out (pre-sigmoid)
    float* __restrict__ v,             // in: v raw -> out: v lerped
    const float* __restrict__ v_first,
    const float* __restrict__ wb, const float* __restrict__ ab,
    const float* __restrict__ vb) {
  long idx = (long)blockIdx.x * blockDim.x + threadIdx.x;
  if (idx >= (long)M_ * H_) return;
  int h = (int)(idx % H_);
  wpre[idx] = DECAY_ * sigm(wpre[idx] + wb[h]);
  apre[idx] = sigm(apre[idx] + ab[h]);
  float vv = v[idx];
  float tt = sigm(vt[idx] + vb[h]);
  v[idx] = vv + tt * (v_first[idx] - vv);
}

// ---------------- kk normalize + k modulation (one wave per head-row) -----
__global__ __launch_bounds__(256) void k_kkprep(
    const float* __restrict__ K, const float* __restrict__ A,
    const float* __restrict__ kkw, const float* __restrict__ kaw,
    float* __restrict__ KK, float* __restrict__ Kmod) {
  int row  = blockIdx.x * 8 + (threadIdx.x >> 5);  // over M_*NH_ rows
  int lane = threadIdx.x & 31;
  int head = row % NH_;
  long base = (long)row * HD_;
  float k1 = K[base + lane],        k2 = K[base + lane + 32];
  float kk1 = k1 * kkw[head * HD_ + lane];
  float kk2 = k2 * kkw[head * HD_ + lane + 32];
  float ss = kk1 * kk1 + kk2 * kk2;
  #pragma unroll
  for (int off = 16; off >= 1; off >>= 1) ss += __shfl_xor(ss, off, 32);
  float inv = 1.0f / fmaxf(sqrtf(ss), 1e-12f);
  KK[base + lane]      = kk1 * inv;
  KK[base + lane + 32] = kk2 * inv;
  float a1 = A[base + lane], a2 = A[base + lane + 32];
  Kmod[base + lane]      = k1 * (1.f + (a1 - 1.f) * kaw[head * HD_ + lane]);
  Kmod[base + lane + 32] = k2 * (1.f + (a2 - 1.f) * kaw[head * HD_ + lane + 32]);
}

// ---------------- sequential RWKV7 recurrence -----------------------------
// One block per (batch,head). Thread j owns state row S[j][0..63] in VGPRs;
// per-step operands broadcast via LDS. b = kk*a, a_in = -kk.
__global__ __launch_bounds__(64, 1) void k_scan(
    const float* __restrict__ R,  const float* __restrict__ Wd,
    const float* __restrict__ Km, const float* __restrict__ V,
    const float* __restrict__ KK, const float* __restrict__ A,
    float* __restrict__ O) {
  const int bh = blockIdx.x;          // 0..B_*NH_-1
  const int b  = bh / NH_;
  const int h  = bh % NH_;
  const int j  = threadIdx.x;         // 0..63 (state row)
  __shared__ float r_s[64], ew_s[64], k_s[64], v_s[64], na_s[64], b_s[64];
  float S[64];
  #pragma unroll
  for (int k = 0; k < 64; k++) S[k] = 0.f;
  const long base0 = ((long)b * T_) * H_ + h * HD_;
  for (int t = 0; t < T_; ++t) {
    const long base = base0 + (long)t * H_;
    r_s[j]  = R[base + j];
    ew_s[j] = __expf(Wd[base + j]);
    k_s[j]  = Km[base + j];
    v_s[j]  = V[base + j];
    float kkv = KK[base + j];
    na_s[j] = -kkv;
    b_s[j]  = kkv * A[base + j];
    __syncthreads();
    float sa = 0.f;
    #pragma unroll
    for (int k = 0; k < 64; k++) sa += S[k] * na_s[k];
    const float vj = v_s[j];
    float o = 0.f;
    #pragma unroll
    for (int k = 0; k < 64; k++) {
      float s = S[k] * ew_s[k] + sa * b_s[k] + vj * k_s[k];
      S[k] = s;
      o += s * r_s[k];
    }
    O[base + j] = o;
    __syncthreads();
  }
}

// ---------------- GroupNorm(head) + correction + gate -> bf16 -------------
__global__ __launch_bounds__(256) void k_gnorm(
    const float* __restrict__ O, const float* __restrict__ R,
    const float* __restrict__ Kmod, const float* __restrict__ V,
    const float* __restrict__ G, const float* __restrict__ rk,
    const float* __restrict__ gnw, const float* __restrict__ gnb,
    u16* __restrict__ Y) {
  int row  = blockIdx.x * 8 + (threadIdx.x >> 5);
  int lane = threadIdx.x & 31;
  int head = row % NH_;
  long base = (long)row * HD_;
  float o1 = O[base + lane], o2 = O[base + lane + 32];
  float s  = o1 + o2, s2 = o1 * o1 + o2 * o2;
  float r1 = R[base + lane], r2 = R[base + lane + 32];
  float k1 = Kmod[base + lane], k2 = Kmod[base + lane + 32];
  float c  = r1 * k1 * rk[head * HD_ + lane] + r2 * k2 * rk[head * HD_ + lane + 32];
  #pragma unroll
  for (int off = 16; off >= 1; off >>= 1) {
    s  += __shfl_xor(s,  off, 32);
    s2 += __shfl_xor(s2, off, 32);
    c  += __shfl_xor(c,  off, 32);
  }
  float mu  = s * (1.f / HD_);
  float var = s2 * (1.f / HD_) - mu * mu;
  float rs  = rsqrtf(var + EPS_);
  int ch1 = head * HD_ + lane, ch2 = ch1 + 32;
  float v1 = V[base + lane], v2 = V[base + lane + 32];
  float g1 = G[base + lane], g2 = G[base + lane + 32];
  Y[base + lane]      = f2bf(((o1 - mu) * rs * gnw[ch1] + gnb[ch1] + c * v1) * g1);
  Y[base + lane + 32] = f2bf(((o2 - mu) * rs * gnw[ch2] + gnb[ch2] + c * v2) * g2);
}

// ---------------- host-side orchestration ---------------------------------
extern "C" void kernel_launch(void* const* d_in, const int* in_sizes, int n_in,
                              void* d_out, int out_size, void* d_ws, size_t ws_size,
                              hipStream_t stream) {
  (void)in_sizes; (void)n_in; (void)out_size; (void)ws_size;
  const float* x      = (const float*)d_in[0];
  const float* vfirst = (const float*)d_in[1];
  const float* m_r = (const float*)d_in[2];
  const float* m_w = (const float*)d_in[3];
  const float* m_k = (const float*)d_in[4];
  const float* m_v = (const float*)d_in[5];
  const float* m_a = (const float*)d_in[6];
  const float* m_g = (const float*)d_in[7];
  const float* k_k = (const float*)d_in[8];
  const float* k_a = (const float*)d_in[9];
  const float* r_k = (const float*)d_in[10];
  const float* W_r = (const float*)d_in[11];
  const float* W_k = (const float*)d_in[12];
  const float* W_v = (const float*)d_in[13];
  const float* W_o = (const float*)d_in[14];
  const float* wA  = (const float*)d_in[15];
  const float* wB  = (const float*)d_in[16];
  const float* wb  = (const float*)d_in[17];
  const float* vA  = (const float*)d_in[18];
  const float* vB  = (const float*)d_in[19];
  const float* vb  = (const float*)d_in[20];
  const float* aA  = (const float*)d_in[21];
  const float* aB  = (const float*)d_in[22];
  const float* ab  = (const float*)d_in[23];
  const float* gA  = (const float*)d_in[24];
  const float* gB  = (const float*)d_in[25];
  const float* gnw = (const float*)d_in[26];
  const float* gnb = (const float*)d_in[27];
  float* out = (float*)d_out;

  const long MH = (long)M_ * H_;
  char* wsp = (char*)d_ws;
  auto alloc = [&](size_t bytes) {
    char* p = wsp;
    wsp += (bytes + 255) & ~(size_t)255;
    return p;
  };
  // bf16 activation matrices
  u16* XR = (u16*)alloc(MH * 2); u16* XW = (u16*)alloc(MH * 2);
  u16* XK = (u16*)alloc(MH * 2); u16* XV = (u16*)alloc(MH * 2);
  u16* XA = (u16*)alloc(MH * 2); u16* XG = (u16*)alloc(MH * 2);
  // bf16 weights
  u16* Wr_b = (u16*)alloc((size_t)H_ * H_ * 2);
  u16* Wk_b = (u16*)alloc((size_t)H_ * H_ * 2);
  u16* Wv_b = (u16*)alloc((size_t)H_ * H_ * 2);
  u16* Wo_b = (u16*)alloc((size_t)H_ * H_ * 2);
  u16* wA_b = (u16*)alloc((size_t)H_ * 64 * 2);
  u16* vA_b = (u16*)alloc((size_t)H_ * 64 * 2);
  u16* aA_b = (u16*)alloc((size_t)H_ * 64 * 2);
  u16* gA_b = (u16*)alloc((size_t)H_ * 160 * 2);
  u16* wB_b = (u16*)alloc((size_t)64 * H_ * 2);
  u16* vB_b = (u16*)alloc((size_t)64 * H_ * 2);
  u16* aB_b = (u16*)alloc((size_t)64 * H_ * 2);
  u16* gB_b = (u16*)alloc((size_t)160 * H_ * 2);
  // f32 intermediates
  float* R    = (float*)alloc(MH * 4);
  float* Kb   = (float*)alloc(MH * 4);
  float* V    = (float*)alloc(MH * 4);   // v raw -> v final (in place)
  float* Wd   = (float*)alloc(MH * 4);   // w pre -> log-decay (in place)
  float* Ap   = (float*)alloc(MH * 4);   // a pre -> a (in place)
  float* Vt   = (float*)alloc(MH * 4);
  float* G    = (float*)alloc(MH * 4);
  float* KK   = (float*)alloc(MH * 4);
  float* Kmod = (float*)alloc(MH * 4);
  float* O    = (float*)alloc(MH * 4);
  float* HW   = (float*)alloc((size_t)M_ * 64 * 4);
  float* HV   = (float*)alloc((size_t)M_ * 64 * 4);
  float* HA   = (float*)alloc((size_t)M_ * 64 * 4);
  float* HG   = (float*)alloc((size_t)M_ * 160 * 4);
  u16* HWb = (u16*)alloc((size_t)M_ * 64 * 2);
  u16* HVb = (u16*)alloc((size_t)M_ * 64 * 2);
  u16* HAb = (u16*)alloc((size_t)M_ * 64 * 2);
  u16* HGb = (u16*)alloc((size_t)M_ * 160 * 2);
  u16* Y   = (u16*)alloc(MH * 2);

  const int EW = 256;
  dim3 blk(EW);

  // 1) token shift + interpolation
  k_token_shift<<<dim3((unsigned)(MH / EW)), blk, 0, stream>>>(
      x, m_r, m_w, m_k, m_v, m_a, m_g, XR, XW, XK, XV, XA, XG);

  // 2) weight conversions
  auto cvt = [&](const float* s, u16* d, long n, int op) {
    k_cvt<<<dim3((unsigned)((n + EW - 1) / EW)), blk, 0, stream>>>(s, d, n, op);
  };
  cvt(W_r, Wr_b, (long)H_ * H_, 0);  cvt(W_k, Wk_b, (long)H_ * H_, 0);
  cvt(W_v, Wv_b, (long)H_ * H_, 0);  cvt(W_o, Wo_b, (long)H_ * H_, 0);
  cvt(wA, wA_b, (long)H_ * 64, 0);   cvt(vA, vA_b, (long)H_ * 64, 0);
  cvt(aA, aA_b, (long)H_ * 64, 0);   cvt(gA, gA_b, (long)H_ * 160, 0);
  cvt(wB, wB_b, (long)64 * H_, 0);   cvt(vB, vB_b, (long)64 * H_, 0);
  cvt(aB, aB_b, (long)64 * H_, 0);   cvt(gB, gB_b, (long)160 * H_, 0);

  // 3) GEMMs (WMMA bf16, TDM-staged A tiles)
  auto gemm = [&](const u16* Am, const u16* Wm, float* Cm, int N, int K) {
    dim3 grid((N + BN - 1) / BN, M_ / BM);
    k_gemm_bf16<<<grid, blk, 0, stream>>>(Am, Wm, Cm, N, K);
  };
  gemm(XR, Wr_b, R,  H_, H_);
  gemm(XK, Wk_b, Kb, H_, H_);
  gemm(XV, Wv_b, V,  H_, H_);
  gemm(XW, wA_b, HW, 64, H_);
  gemm(XV, vA_b, HV, 64, H_);
  gemm(XA, aA_b, HA, 64, H_);
  gemm(XG, gA_b, HG, 160, H_);

  // 4) LoRA mid activations -> bf16
  cvt(HW, HWb, (long)M_ * 64, 1);    // tanh
  cvt(HV, HVb, (long)M_ * 64, 0);
  cvt(HA, HAb, (long)M_ * 64, 0);
  cvt(HG, HGb, (long)M_ * 160, 2);   // sigmoid

  // 5) LoRA up GEMMs
  gemm(HWb, wB_b, Wd, H_, 64);
  gemm(HVb, vB_b, Vt, H_, 64);
  gemm(HAb, aB_b, Ap, H_, 64);
  gemm(HGb, gB_b, G,  H_, 160);

  // 6) elementwise post: w, a, v
  k_post<<<dim3((unsigned)(MH / EW)), blk, 0, stream>>>(
      Wd, Ap, Vt, V, vfirst, wb, ab, vb);

  // 7) kk normalize + k modulation
  k_kkprep<<<dim3((unsigned)(M_ * NH_ / 8)), blk, 0, stream>>>(
      Kb, Ap, k_k, k_a, KK, Kmod);

  // 8) sequential recurrence
  k_scan<<<dim3(B_ * NH_), dim3(64), 0, stream>>>(R, Wd, Kmod, V, KK, Ap, O);

  // 9) groupnorm + correction + gate -> bf16 Y
  k_gnorm<<<dim3((unsigned)(M_ * NH_ / 8)), blk, 0, stream>>>(
      O, R, Kmod, V, G, r_k, gnw, gnb, Y);

  // 10) output projection
  gemm(Y, Wo_b, out, H_, H_);
}
